// CausalMambaGCN_6854767804967
// MI455X (gfx1250) — matrix operand order repaired and checked by reading, not compiled, not run on hardware
//
#include <hip/hip_runtime.h>
#include <math.h>

// ---------------- problem constants ----------------
#define NN 8192
#define DD 768
#define EE 262144
#define DI 1536
#define DS 16
#define DCONV 4
#define DTR 48
#define DBC (DTR + 2*DS)   // 80

// ---------------- WMMA types ----------------
typedef __attribute__((ext_vector_type(16))) __bf16 v16bf;
typedef __attribute__((ext_vector_type(8)))  float  v8f;

union FragB16 { v16bf bf; unsigned int u32[8]; };

__device__ inline unsigned short f32_to_bf16(float f) {
    unsigned int u = __float_as_uint(f);
    unsigned int r = 0x7FFFu + ((u >> 16) & 1u);
    return (unsigned short)((u + r) >> 16);
}

__device__ inline float siluf(float x) { return x / (1.f + __expf(-x)); }
__device__ inline float geluf(float x) { return 0.5f * x * (1.f + erff(x * 0.70710678118654752f)); }
__device__ inline float softplusf(float x) { return fmaxf(x, 0.f) + log1pf(__expf(-fabsf(x))); }

// ---------------- generic bf16 WMMA GEMM ----------------
// C[M,N] = act( A[M,Kpad](f32, lda, optional row-gather) @ Bt + bias[N] )
// Bt: bf16 weights stored TRANSPOSED + PADDED: Bt[n*Kpad + k], zero outside [K,N].
// Requirements: M % 128 == 0, Kpad % 32 == 0, A rows have >= Kpad readable floats,
//               Bt has >= ceil(N/128)*128 rows.
// act: 0=none, 1=gelu, 2=softplus
#define BM 128
#define BN 128
#define KT 32
#define APAD 40
#define BPAD 40

template <bool GATHER>
__global__ __launch_bounds__(256)
void cmg_gemm_bf16(const float* __restrict__ A, int lda, const int* __restrict__ gather,
                   const unsigned short* __restrict__ Bt, int Kpad,
                   const float* __restrict__ bias,
                   float* __restrict__ C, int M, int N, int act)
{
    __shared__ unsigned short As[2][BM][APAD];
    __shared__ unsigned short Bs[2][BN][BPAD];

    const int tid  = threadIdx.x;
    const int lane = tid & 31;
    const int wid  = tid >> 5;
    const int waveM = wid & 3;       // 4 groups of 32 rows
    const int waveN = wid >> 2;      // 2 groups of 64 cols
    const int bm0 = blockIdx.y * BM;
    const int bn0 = blockIdx.x * BN;
    const int row16 = lane & 15;
    const int half  = lane >> 4;

    // hoisted, k-invariant A row pointers (gather resolved once)
    const float* aptr[4];
    int arow[4], akk[4];
    #pragma unroll
    for (int i = 0; i < 4; ++i) {
        int e = (tid + i * 256) * 4;      // 0..4092
        int r = e >> 5;                   // 0..127
        int kk = e & 31;                  // 0,4,..,28
        arow[i] = r; akk[i] = kk;
        int gr = bm0 + r;
        int ar = GATHER ? gather[gr] : gr;
        aptr[i] = A + (size_t)ar * lda + kk;
    }
    // hoisted, k-invariant B global row bases (chunks of 4 ushorts)
    const unsigned short* bptr[4];
    int brow[4], bkk[4];
    #pragma unroll
    for (int i = 0; i < 4; ++i) {
        int idx = tid + i * 256;          // 0..1023 chunks
        int n  = idx >> 3;                // 0..127
        int kk = (idx & 7) * 4;           // 0,4,..,28
        brow[i] = n; bkk[i] = kk;
        bptr[i] = Bt + (size_t)(bn0 + n) * (size_t)Kpad + kk;
    }

    auto stageB = [&](int k0, int buf) {
        #pragma unroll
        for (int i = 0; i < 4; ++i) {
            unsigned int ldsoff = (unsigned int)(reinterpret_cast<uintptr_t>(&Bs[buf][brow[i]][bkk[i]]));
            unsigned long long ga = reinterpret_cast<unsigned long long>(bptr[i] + k0);
            asm volatile("global_load_async_to_lds_b64 %0, %1, off"
                         :: "v"(ldsoff), "v"(ga) : "memory");
        }
    };
    auto stageA = [&](int k0, int buf) {
        #pragma unroll
        for (int i = 0; i < 4; ++i) {
            float4 v = *(const float4*)(aptr[i] + k0);
            unsigned int p0 = (unsigned int)f32_to_bf16(v.x) | ((unsigned int)f32_to_bf16(v.y) << 16);
            unsigned int p1 = (unsigned int)f32_to_bf16(v.z) | ((unsigned int)f32_to_bf16(v.w) << 16);
            uint2 pk; pk.x = p0; pk.y = p1;
            *(uint2*)&As[buf][arow[i]][akk[i]] = pk;
        }
    };

    v8f acc[2][4] = {};

    // prologue: stage tile 0
    stageB(0, 0);
    stageA(0, 0);
    asm volatile("s_wait_asynccnt 0x0" ::: "memory");
    __syncthreads();

    int buf = 0;
    for (int k0 = 0; k0 < Kpad; k0 += KT) {
        int nk = k0 + KT;
        // stage next tile into the other buffer while computing this one
        if (nk < Kpad) {
            stageB(nk, buf ^ 1);
            stageA(nk, buf ^ 1);
        }

        FragB16 fa[2], fb[4];
        #pragma unroll
        for (int im = 0; im < 2; ++im) {
            int row = waveM * 32 + im * 16 + row16;
            #pragma unroll
            for (int v = 0; v < 8; ++v) {
                int kb = ((v < 4) ? 0 : 16) + half * 8 + (v & 3) * 2;   // ISA A-16bit layout
                fa[im].u32[v] = *(const unsigned int*)&As[buf][row][kb];
            }
        }
        #pragma unroll
        for (int in = 0; in < 4; ++in) {
            int col = waveN * 64 + in * 16 + row16;
            #pragma unroll
            for (int v = 0; v < 8; ++v) {
                int kb = ((v < 4) ? 0 : 16) + half * 8 + (v & 3) * 2;   // same k order as A
                fb[in].u32[v] = *(const unsigned int*)&Bs[buf][col][kb];
            }
        }
        #pragma unroll
        for (int im = 0; im < 2; ++im)
            #pragma unroll
            for (int in = 0; in < 4; ++in)
                acc[im][in] = __builtin_amdgcn_wmma_f32_16x16x32_bf16(
                    false, fa[im].bf, false, fb[in].bf,
                    (short)0, acc[im][in], false, false);

        asm volatile("s_wait_asynccnt 0x0" ::: "memory");
        __syncthreads();
        buf ^= 1;
    }

    // store C (lane = col, vgpr v = row M = v + 8*half); M always tile-aligned
    #pragma unroll
    for (int im = 0; im < 2; ++im) {
        #pragma unroll
        for (int in = 0; in < 4; ++in) {
            int col = bn0 + waveN * 64 + in * 16 + row16;
            if (col >= N) continue;
            float bv = bias ? bias[col] : 0.f;
            #pragma unroll
            for (int v = 0; v < 8; ++v) {
                int row = bm0 + waveM * 32 + im * 16 + v + 8 * half;
                float x = acc[im][in][v] + bv;
                if (act == 1) x = geluf(x);
                else if (act == 2) x = softplusf(x);
                C[(size_t)row * N + col] = x;
            }
        }
    }
}

// ---------------- small kernels ----------------
// weights f32 [K,N] -> bf16 transposed padded [Npad, Kpad], zero-filled
__global__ void cmg_convert_bf16_t(const float* __restrict__ in, unsigned short* __restrict__ out,
                                   int K, int N, int Kpad, int Npad) {
    int idx = blockIdx.x * blockDim.x + threadIdx.x;
    if (idx >= Npad * Kpad) return;
    int n = idx / Kpad, k = idx % Kpad;
    float v = (n < N && k < K) ? in[(size_t)k * N + n] : 0.f;
    out[idx] = f32_to_bf16(v);
}

__global__ void cmg_deg(const int* __restrict__ src, const int* __restrict__ dst,
                        float* __restrict__ outdeg, float* __restrict__ indeg) {
    int e = blockIdx.x * blockDim.x + threadIdx.x;
    if (e < EE) {
        atomicAdd(&outdeg[src[e]], 1.f);
        atomicAdd(&indeg[dst[e]], 1.f);
    }
}

__global__ void cmg_norm(float* __restrict__ outdeg, float* __restrict__ indeg) {
    int i = blockIdx.x * blockDim.x + threadIdx.x;
    if (i < NN) {
        outdeg[i] = rsqrtf(fmaxf(outdeg[i], 1.f));   // becomes norm_s
        indeg[i]  = rsqrtf(fmaxf(indeg[i], 1.f));    // becomes norm_d
    }
}

// stable argsort by rank counting (matches stable jnp.argsort)
__global__ void cmg_argsort(const int* __restrict__ ts, int* __restrict__ sorted_idx) {
    __shared__ int tsl[256];
    int i = blockIdx.x * blockDim.x + threadIdx.x;
    int ti = ts[i];
    int rank = 0;
    for (int base = 0; base < NN; base += 256) {
        __syncthreads();
        tsl[threadIdx.x] = ts[base + threadIdx.x];
        __syncthreads();
        #pragma unroll 8
        for (int j = 0; j < 256; ++j) {
            int tj = tsl[j];
            int jg = base + j;
            rank += (tj < ti || (tj == ti && jg < i)) ? 1 : 0;
        }
    }
    sorted_idx[rank] = i;
}

// depthwise causal conv over time + SiLU; xi = xz[:, 0:DI]
__global__ void cmg_conv_silu(const float* __restrict__ xz, const float* __restrict__ conv_w,
                              const float* __restrict__ conv_b, float* __restrict__ u) {
    int idx = blockIdx.x * blockDim.x + threadIdx.x;
    if (idx >= NN * DI) return;
    int t = idx / DI, c = idx % DI;
    float acc = conv_b[c];
    #pragma unroll
    for (int k = 0; k < DCONV; ++k) {
        int tt = t + k - (DCONV - 1);
        float xv = (tt >= 0) ? xz[(size_t)tt * (2 * DI) + c] : 0.f;
        acc = fmaf(xv, conv_w[c * DCONV + k], acc);
    }
    u[idx] = siluf(acc);
}

// selective scan: one lane per channel, 16-wide state in registers
__global__ void cmg_scan(const float* __restrict__ delta, const float* __restrict__ dbc,
                         const float* __restrict__ u, const float* __restrict__ A_log,
                         float* __restrict__ y) {
    int ch = blockIdx.x * blockDim.x + threadIdx.x;
    if (ch >= DI) return;
    float a[DS], h[DS];
    #pragma unroll
    for (int s = 0; s < DS; ++s) { a[s] = -__expf(A_log[ch * DS + s]); h[s] = 0.f; }
    for (int t = 0; t < NN; ++t) {
        float dt = delta[(size_t)t * DI + ch];
        float ut = u[(size_t)t * DI + ch];
        float du = dt * ut;
        const float* bc = dbc + (size_t)t * DBC;
        float acc = 0.f;
        #pragma unroll
        for (int s = 0; s < DS; ++s) {
            float dA = __expf(dt * a[s]);
            h[s] = fmaf(h[s], dA, du * bc[DTR + s]);
            acc  = fmaf(h[s], bc[DTR + DS + s], acc);
        }
        y[(size_t)t * DI + ch] = acc;
    }
}

// y = (y + u*D) * silu(z);  z = xz[:, DI:2*DI]
__global__ void cmg_gate(float* __restrict__ y, const float* __restrict__ u,
                         const float* __restrict__ xz, const float* __restrict__ D_param) {
    int idx = blockIdx.x * blockDim.x + threadIdx.x;
    if (idx >= NN * DI) return;
    int t = idx / DI, c = idx % DI;
    float z = xz[(size_t)t * (2 * DI) + DI + c];
    y[idx] = (y[idx] + u[idx] * D_param[c]) * siluf(z);
}

// edge-parallel scatter-add: m[dst] += h[src] * norm_s[src]
__global__ void cmg_scatter(const float* __restrict__ h, const int* __restrict__ src,
                            const int* __restrict__ dst, const float* __restrict__ norm_s,
                            float* __restrict__ m) {
    int e = blockIdx.x;
    int s = src[e], d = dst[e];
    float ns = norm_s[s];
    const float* hp = h + (size_t)s * DD;
    float* mp = m + (size_t)d * DD;
    for (int f = threadIdx.x; f < DD; f += blockDim.x)
        atomicAdd(&mp[f], hp[f] * ns);
}

__global__ void cmg_rowscale(float* __restrict__ m, const float* __restrict__ norm_d) {
    int idx = blockIdx.x * blockDim.x + threadIdx.x;
    if (idx >= NN * DD) return;
    m[idx] *= norm_d[idx / DD];
}

// causal = sigmoid(h @ W_c + b_c); flag = causal > 0.7 ; one wave per row
__global__ void cmg_causal(const float* __restrict__ h, const float* __restrict__ W_c,
                           const float* __restrict__ b_c,
                           float* __restrict__ out_causal, float* __restrict__ out_flag) {
    int lane = threadIdx.x & 31;
    int row = blockIdx.x * (blockDim.x >> 5) + (threadIdx.x >> 5);
    const float* hp = h + (size_t)row * DD;
    float sum = 0.f;
    for (int f = lane; f < DD; f += 32) sum = fmaf(hp[f], W_c[f], sum);
    #pragma unroll
    for (int off = 16; off > 0; off >>= 1) sum += __shfl_xor(sum, off, 32);
    if (lane == 0) {
        float c = 1.f / (1.f + __expf(-(sum + b_c[0])));
        out_causal[row] = c;
        out_flag[row] = (c > 0.7f) ? 1.f : 0.f;
    }
}

__global__ void cmg_colmean(const float* __restrict__ h, float* __restrict__ out) {
    int c = blockIdx.x * blockDim.x + threadIdx.x;
    if (c >= DD) return;
    float sum = 0.f;
    for (int t = 0; t < NN; ++t) sum += h[(size_t)t * DD + c];
    out[c] = sum * (1.f / (float)NN);
}

// ---------------- launcher ----------------
extern "C" void kernel_launch(void* const* d_in, const int* in_sizes, int n_in,
                              void* d_out, int out_size, void* d_ws, size_t ws_size,
                              hipStream_t stream) {
    const float* node_feats = (const float*)d_in[0];
    const int*   timestamps = (const int*)d_in[1];
    const int*   edge_index = (const int*)d_in[2];
    const float* W_in    = (const float*)d_in[3];
    const float* conv_w  = (const float*)d_in[4];
    const float* conv_b  = (const float*)d_in[5];
    const float* W_xproj = (const float*)d_in[6];
    const float* W_dt    = (const float*)d_in[7];
    const float* b_dt    = (const float*)d_in[8];
    const float* A_log   = (const float*)d_in[9];
    const float* D_param = (const float*)d_in[10];
    const float* W_out   = (const float*)d_in[11];
    const float* Wg      = (const float*)d_in[12];
    const float* bg      = (const float*)d_in[13];
    const float* W_c     = (const float*)d_in[14];
    const float* b_c     = (const float*)d_in[15];

    const int* src = edge_index;        // edge_index[0,:]
    const int* dst = edge_index + EE;   // edge_index[1,:]

    // padded bf16 weight shapes (transposed [Npad, Kpad])
    const int KP_dt = 64;      // K=48 -> 64
    const int NP_xp = 128;     // N=80 -> 128

    // workspace bump allocator
    char* base = (char*)d_ws;
    size_t off = 0;
    auto alloc = [&](size_t bytes) -> void* {
        void* p = base + off;
        off = (off + bytes + 255) & ~(size_t)255;
        return p;
    };
    int*   sorted_idx = (int*)  alloc(NN * sizeof(int));
    float* norm_s     = (float*)alloc(NN * sizeof(float));
    float* norm_d     = (float*)alloc(NN * sizeof(float));
    float* xz         = (float*)alloc((size_t)NN * 2 * DI * sizeof(float));
    float* u          = (float*)alloc((size_t)NN * DI * sizeof(float));
    float* dbc        = (float*)alloc((size_t)NN * DBC * sizeof(float));
    float* delta      = (float*)alloc((size_t)NN * DI * sizeof(float));
    float* y          = (float*)alloc((size_t)NN * DI * sizeof(float));
    float* hA         = (float*)alloc((size_t)NN * DD * sizeof(float));
    float* hB         = (float*)alloc((size_t)NN * DD * sizeof(float));
    float* mbuf       = (float*)alloc((size_t)NN * DD * sizeof(float));
    unsigned short* W_in_bf    = (unsigned short*)alloc((size_t)(2 * DI) * DD * sizeof(unsigned short));      // [3072][768]
    unsigned short* W_xproj_bf = (unsigned short*)alloc((size_t)NP_xp * DI * sizeof(unsigned short));         // [128][1536]
    unsigned short* W_dt_bf    = (unsigned short*)alloc((size_t)DI * KP_dt * sizeof(unsigned short));         // [1536][64]
    unsigned short* W_out_bf   = (unsigned short*)alloc((size_t)DD * DI * sizeof(unsigned short));            // [768][1536]
    unsigned short* Wg_bf      = (unsigned short*)alloc((size_t)3 * DD * DD * sizeof(unsigned short));        // 3x[768][768]
    if (off > ws_size) return;

    // 1) degree norms
    hipMemsetAsync(norm_s, 0, NN * sizeof(float), stream);
    hipMemsetAsync(norm_d, 0, NN * sizeof(float), stream);
    cmg_deg<<<(EE + 255) / 256, 256, 0, stream>>>(src, dst, norm_s, norm_d);
    cmg_norm<<<(NN + 255) / 256, 256, 0, stream>>>(norm_s, norm_d);

    // 2) stable argsort of timestamps
    cmg_argsort<<<NN / 256, 256, 0, stream>>>(timestamps, sorted_idx);

    // 3) convert weights -> bf16, transposed, padded
    cmg_convert_bf16_t<<<((2 * DI * DD) + 255) / 256, 256, 0, stream>>>(W_in, W_in_bf, DD, 2 * DI, DD, 2 * DI);
    cmg_convert_bf16_t<<<((NP_xp * DI) + 255) / 256, 256, 0, stream>>>(W_xproj, W_xproj_bf, DI, DBC, DI, NP_xp);
    cmg_convert_bf16_t<<<((DI * KP_dt) + 255) / 256, 256, 0, stream>>>(W_dt, W_dt_bf, DTR, DI, KP_dt, DI);
    cmg_convert_bf16_t<<<((DD * DI) + 255) / 256, 256, 0, stream>>>(W_out, W_out_bf, DI, DD, DI, DD);
    for (int i = 0; i < 3; ++i)
        cmg_convert_bf16_t<<<((DD * DD) + 255) / 256, 256, 0, stream>>>(
            Wg + (size_t)i * DD * DD, Wg_bf + (size_t)i * DD * DD, DD, DD, DD, DD);

    // 4) xz = seq @ W_in  (gathered rows), M=8192 K=768 N=3072
    {
        dim3 g((2 * DI) / BN, NN / BM);
        cmg_gemm_bf16<true><<<g, 256, 0, stream>>>(node_feats, DD, sorted_idx, W_in_bf, DD,
                                                   nullptr, xz, NN, 2 * DI, 0);
    }
    // 5) u = silu(causal depthwise conv(xi))
    cmg_conv_silu<<<(NN * DI + 255) / 256, 256, 0, stream>>>(xz, conv_w, conv_b, u);

    // 6) dbc = u @ W_xproj, M=8192 K=1536 N=80 (Npad=128)
    {
        dim3 g(NP_xp / BN, NN / BM);
        cmg_gemm_bf16<false><<<g, 256, 0, stream>>>(u, DI, nullptr, W_xproj_bf, DI,
                                                    nullptr, dbc, NN, DBC, 0);
    }
    // 7) delta = softplus(dbc[:, :48] @ W_dt + b_dt), M=8192 K=48 (Kpad=64) N=1536
    {
        dim3 g(DI / BN, NN / BM);
        cmg_gemm_bf16<false><<<g, 256, 0, stream>>>(dbc, DBC, nullptr, W_dt_bf, KP_dt,
                                                    b_dt, delta, NN, DI, 2);
    }
    // 8) selective scan -> y
    cmg_scan<<<DI / 256, 256, 0, stream>>>(delta, dbc, u, A_log, y);

    // 9) gate: y = (y + u*D) * silu(z)
    cmg_gate<<<(NN * DI + 255) / 256, 256, 0, stream>>>(y, u, xz, D_param);

    // 10) h0 = y @ W_out, M=8192 K=1536 N=768
    {
        dim3 g(DD / BN, NN / BM);
        cmg_gemm_bf16<false><<<g, 256, 0, stream>>>(y, DI, nullptr, W_out_bf, DI,
                                                    nullptr, hA, NN, DD, 0);
    }

    // 11) three GCN layers with GELU
    float* hcur = hA;
    float* hnext = hB;
    for (int i = 0; i < 3; ++i) {
        hipMemsetAsync(mbuf, 0, (size_t)NN * DD * sizeof(float), stream);
        cmg_scatter<<<EE, 256, 0, stream>>>(hcur, src, dst, norm_s, mbuf);
        cmg_rowscale<<<(NN * DD + 255) / 256, 256, 0, stream>>>(mbuf, norm_d);
        dim3 g(DD / BN, NN / BM);
        cmg_gemm_bf16<false><<<g, 256, 0, stream>>>(mbuf, DD, nullptr,
                                                    Wg_bf + (size_t)i * DD * DD, DD,
                                                    bg + (size_t)i * DD, hnext, NN, DD, 1);
        float* t = hcur; hcur = hnext; hnext = t;
    }

    // 12) outputs: [0:768] graph_feat, [768:8960] causal, [8960:17152] flag
    float* out = (float*)d_out;
    cmg_colmean<<<(DD + 255) / 256, 256, 0, stream>>>(hcur, out);
    cmg_causal<<<NN / 8, 256, 0, stream>>>(hcur, W_c, b_c, out + DD, out + DD + NN);
}
